// HeadAttention_44736379355516
// MI455X (gfx1250) — matrix operand order repaired and checked by reading, not compiled, not run on hardware
//
#include <hip/hip_runtime.h>

typedef __attribute__((ext_vector_type(16))) _Float16 v16h;
typedef __attribute__((ext_vector_type(8)))  _Float16 v8h;
typedef __attribute__((ext_vector_type(8)))  float    v8f;
typedef __attribute__((ext_vector_type(4)))  float    v4f;

#define SEQ   2048
#define EMB   1024
#define HID   1024
#define BATCH 8
#define ROWS  (BATCH * SEQ)   // 16384 flattened (b,s) rows

union V16 { v16h v; v8h h[2]; };

#define WMMA_F16(A, B, C) \
  __builtin_amdgcn_wmma_f32_16x16x32_f16(false, (A).v, false, (B).v, (short)0, (C), false, false)

// ---------------------------------------------------------------------------
// Prep: Wt[h][e] = (f16) W[e][h]   for Wq, Wk, Wv  (grid.y selects matrix)
// ---------------------------------------------------------------------------
__global__ void ha_wt_kernel(const float* __restrict__ w0, const float* __restrict__ w1,
                             const float* __restrict__ w2,
                             _Float16* __restrict__ t0, _Float16* __restrict__ t1,
                             _Float16* __restrict__ t2) {
  const float* w = (blockIdx.y == 0) ? w0 : ((blockIdx.y == 1) ? w1 : w2);
  _Float16*    t = (blockIdx.y == 0) ? t0 : ((blockIdx.y == 1) ? t1 : t2);
  int idx = blockIdx.x * 256 + threadIdx.x;   // idx = h*EMB + e
  int h = idx >> 10;
  int e = idx & 1023;
  t[idx] = (_Float16)w[e * HID + h];
}

// ---------------------------------------------------------------------------
// Fused QKV projection. One wave = 16 rows x 32 cols for q, k AND v:
// one fp32 A load+convert feeds 6 WMMAs.  q,k row-major f16 [ROWS][HID];
// v stored transposed Vt[b][h][s] f16 (b128 packed stores via C layout).
// ---------------------------------------------------------------------------
__global__ void ha_qkv_kernel(const float* __restrict__ x,
                              const _Float16* __restrict__ wtq,
                              const _Float16* __restrict__ wtk,
                              const _Float16* __restrict__ wtv,
                              const float* __restrict__ bq,
                              const float* __restrict__ bk,
                              const float* __restrict__ bv,
                              _Float16* __restrict__ q,
                              _Float16* __restrict__ k,
                              _Float16* __restrict__ vt) {
  const int wave  = threadIdx.x >> 5;
  const int lane  = threadIdx.x & 31;
  const int tile  = blockIdx.x * 8 + wave;
  const int mtile = tile >> 5;                 // 1024 m-tiles (16 rows)
  const int npair = tile & 31;                 // 32 n-pairs (32 cols)
  const int half  = (lane >> 4) & 1;
  const int l15   = lane & 15;

  const float*    ap   = x + (size_t)(mtile * 16 + l15) * EMB + half * 16;
  const size_t    brow = (size_t)(npair * 32 + l15) * EMB + half * 16;
  const _Float16* bqp0 = wtq + brow;  const _Float16* bqp1 = bqp0 + (size_t)16 * EMB;
  const _Float16* bkp0 = wtk + brow;  const _Float16* bkp1 = bkp0 + (size_t)16 * EMB;
  const _Float16* bvp0 = wtv + brow;  const _Float16* bvp1 = bvp0 + (size_t)16 * EMB;

  v8f cq0 = {}, cq1 = {}, ck0 = {}, ck1 = {}, cv0 = {}, cv1 = {};
  for (int kk = 0; kk < EMB; kk += 32) {
    V16 a;
    v4f f0 = *(const v4f*)(ap + kk);
    v4f f1 = *(const v4f*)(ap + kk + 4);
    v4f f2 = *(const v4f*)(ap + kk + 8);
    v4f f3 = *(const v4f*)(ap + kk + 12);
#pragma unroll
    for (int i = 0; i < 4; ++i) {
      a.v[i]      = (_Float16)f0[i];
      a.v[4 + i]  = (_Float16)f1[i];
      a.v[8 + i]  = (_Float16)f2[i];
      a.v[12 + i] = (_Float16)f3[i];
    }
    V16 b0, b1;
    b0.h[0] = *(const v8h*)(bqp0 + kk); b0.h[1] = *(const v8h*)(bqp0 + kk + 8);
    b1.h[0] = *(const v8h*)(bqp1 + kk); b1.h[1] = *(const v8h*)(bqp1 + kk + 8);
    cq0 = WMMA_F16(a, b0, cq0);         cq1 = WMMA_F16(a, b1, cq1);
    b0.h[0] = *(const v8h*)(bkp0 + kk); b0.h[1] = *(const v8h*)(bkp0 + kk + 8);
    b1.h[0] = *(const v8h*)(bkp1 + kk); b1.h[1] = *(const v8h*)(bkp1 + kk + 8);
    ck0 = WMMA_F16(a, b0, ck0);         ck1 = WMMA_F16(a, b1, ck1);
    b0.h[0] = *(const v8h*)(bvp0 + kk); b0.h[1] = *(const v8h*)(bvp0 + kk + 8);
    b1.h[0] = *(const v8h*)(bvp1 + kk); b1.h[1] = *(const v8h*)(bvp1 + kk + 8);
    cv0 = WMMA_F16(a, b0, cv0);         cv1 = WMMA_F16(a, b1, cv1);
  }

  const int ncol0 = npair * 32 + l15;
  const int ncol1 = ncol0 + 16;
  const int rbase = mtile * 16 + half * 8;     // C layout: VGPR r -> row rbase+r
  const float biq0 = bq[ncol0], biq1 = bq[ncol1];
  const float bik0 = bk[ncol0], bik1 = bk[ncol1];
  const float biv0 = bv[ncol0], biv1 = bv[ncol1];

#pragma unroll
  for (int r = 0; r < 8; ++r) {
    const size_t row = (size_t)(rbase + r) * HID;
    q[row + ncol0] = (_Float16)(cq0[r] + biq0);
    q[row + ncol1] = (_Float16)(cq1[r] + biq1);
    k[row + ncol0] = (_Float16)(ck0[r] + bik0);
    k[row + ncol1] = (_Float16)(ck1[r] + bik1);
  }
  // V transposed: lane's column fixed, its 8 rows contiguous -> one b128 store
  const int bidx  = rbase >> 11;      // /SEQ
  const int sbase = rbase & 2047;     // %SEQ (16-row tiles never cross batch)
  v8h vv0, vv1;
#pragma unroll
  for (int r = 0; r < 8; ++r) {
    vv0[r] = (_Float16)(cv0[r] + biv0);
    vv1[r] = (_Float16)(cv1[r] + biv1);
  }
  _Float16* vbase = vt + (size_t)bidx * HID * SEQ + sbase;
  *(v8h*)(vbase + (size_t)ncol0 * SEQ) = vv0;
  *(v8h*)(vbase + (size_t)ncol1 * SEQ) = vv1;
}

// ---------------------------------------------------------------------------
// Causal score blocks for batch b: 32x32 per wave (2x2 WMMA tiles, 4 WMMA per
// 8 loads).  S[i][j] = (q_i . k_j)/32 for j<=i else -inf.  Blocks fully above
// the diagonal are skipped and never read back.
// ---------------------------------------------------------------------------
__device__ inline void ha_store_score(float* __restrict__ s, const v8f& c,
                                      int ibase, int j) {
#pragma unroll
  for (int r = 0; r < 8; ++r) {
    const int i = ibase + r;
    s[(size_t)i * SEQ + j] = (j <= i) ? (c[r] * 0.03125f) : -__builtin_inff();
  }
}

__global__ void ha_scores_kernel(const _Float16* __restrict__ q,
                                 const _Float16* __restrict__ k,
                                 float* __restrict__ s, int b) {
  const int wave = threadIdx.x >> 5;
  const int lane = threadIdx.x & 31;
  const int jt   = blockIdx.x * 8 + wave;   // 64 j-blocks of 32
  const int it   = blockIdx.y;              // 64 i-blocks of 32
  if (jt * 32 > it * 32 + 31) return;       // strictly upper block: skip

  const int half = lane >> 4;
  const int l15  = lane & 15;
  const _Float16* ap0 = q + (size_t)(b * SEQ + it * 32 + l15) * HID + half * 16;
  const _Float16* ap1 = ap0 + (size_t)16 * HID;
  const _Float16* bp0 = k + (size_t)(b * SEQ + jt * 32 + l15) * HID + half * 16;
  const _Float16* bp1 = bp0 + (size_t)16 * HID;

  v8f c00 = {}, c01 = {}, c10 = {}, c11 = {};
#pragma unroll 2
  for (int kk = 0; kk < HID; kk += 32) {
    V16 a0, a1, b0, b1;
    a0.h[0] = *(const v8h*)(ap0 + kk); a0.h[1] = *(const v8h*)(ap0 + kk + 8);
    a1.h[0] = *(const v8h*)(ap1 + kk); a1.h[1] = *(const v8h*)(ap1 + kk + 8);
    b0.h[0] = *(const v8h*)(bp0 + kk); b0.h[1] = *(const v8h*)(bp0 + kk + 8);
    b1.h[0] = *(const v8h*)(bp1 + kk); b1.h[1] = *(const v8h*)(bp1 + kk + 8);
    c00 = WMMA_F16(a0, b0, c00);  c01 = WMMA_F16(a0, b1, c01);
    c10 = WMMA_F16(a1, b0, c10);  c11 = WMMA_F16(a1, b1, c11);
  }

  const int ib = it * 32 + half * 8;
  const int jb = jt * 32 + l15;
  ha_store_score(s, c00, ib,      jb);
  ha_store_score(s, c01, ib,      jb + 16);
  ha_store_score(s, c10, ib + 16, jb);
  ha_store_score(s, c11, ib + 16, jb + 16);
}

// ---------------------------------------------------------------------------
// Row softmax (f32 in, f16 out). One block per query row; zeros written past
// the causal limit so the AV GEMM can read aligned 32-wide K chunks.
// ---------------------------------------------------------------------------
__global__ void ha_softmax_kernel(const float* __restrict__ s, _Float16* __restrict__ p) {
  const int i   = blockIdx.x;
  const int len = i + 1;
  const int t   = threadIdx.x;
  __shared__ float red[256];

  float m = -__builtin_inff();
  for (int j = t; j < len; j += 256) m = fmaxf(m, s[(size_t)i * SEQ + j]);
  red[t] = m; __syncthreads();
  for (int off = 128; off; off >>= 1) {
    if (t < off) red[t] = fmaxf(red[t], red[t + off]);
    __syncthreads();
  }
  m = red[0]; __syncthreads();

  float sum = 0.0f;
  for (int j = t; j < len; j += 256) sum += __expf(s[(size_t)i * SEQ + j] - m);
  red[t] = sum; __syncthreads();
  for (int off = 128; off; off >>= 1) {
    if (t < off) red[t] += red[t + off];
    __syncthreads();
  }
  const float inv = 1.0f / red[0];

  for (int j = t; j < SEQ; j += 256) {
    float val = (j < len) ? __expf(s[(size_t)i * SEQ + j] - m) * inv : 0.0f;
    p[(size_t)i * SEQ + j] = (_Float16)val;
  }
}

// ---------------------------------------------------------------------------
// out[b] = P x V.  32x32 per wave (2x2 WMMA tiles).  A = P rows (row-major),
// B = Vt columns (contiguous K).  K loop truncated at each block's causal
// limit; P holds exact zeros beyond it.
// ---------------------------------------------------------------------------
__global__ void ha_av_kernel(const _Float16* __restrict__ p,
                             const _Float16* __restrict__ vt,
                             float* __restrict__ out, int b) {
  const int wave = threadIdx.x >> 5;
  const int lane = threadIdx.x & 31;
  const int tile = blockIdx.x * 8 + wave;
  const int it   = tile >> 5;               // 64 i-blocks of 32
  const int ht   = tile & 31;               // 32 h-blocks of 32
  const int half = lane >> 4;
  const int l15  = lane & 15;

  const _Float16* ap0 = p + (size_t)(it * 32 + l15) * SEQ + half * 16;
  const _Float16* ap1 = ap0 + (size_t)16 * SEQ;
  const _Float16* bp0 = vt + (size_t)b * HID * SEQ + (size_t)(ht * 32 + l15) * SEQ + half * 16;
  const _Float16* bp1 = bp0 + (size_t)16 * SEQ;

  v8f c00 = {}, c01 = {}, c10 = {}, c11 = {};
  const int kend = it * 32 + 32;            // keys beyond this are exact zeros
#pragma unroll 2
  for (int kk = 0; kk < kend; kk += 32) {
    V16 a0, a1, b0, b1;
    a0.h[0] = *(const v8h*)(ap0 + kk); a0.h[1] = *(const v8h*)(ap0 + kk + 8);
    a1.h[0] = *(const v8h*)(ap1 + kk); a1.h[1] = *(const v8h*)(ap1 + kk + 8);
    b0.h[0] = *(const v8h*)(bp0 + kk); b0.h[1] = *(const v8h*)(bp0 + kk + 8);
    b1.h[0] = *(const v8h*)(bp1 + kk); b1.h[1] = *(const v8h*)(bp1 + kk + 8);
    c00 = WMMA_F16(a0, b0, c00);  c01 = WMMA_F16(a0, b1, c01);
    c10 = WMMA_F16(a1, b0, c10);  c11 = WMMA_F16(a1, b1, c11);
  }

  float* ob = out + (size_t)b * SEQ * HID;
  const int ib = it * 32 + half * 8;
  const int hb = ht * 32 + l15;
#pragma unroll
  for (int r = 0; r < 8; ++r) {
    ob[(size_t)(ib + r) * HID + hb]           = c00[r];
    ob[(size_t)(ib + r) * HID + hb + 16]      = c01[r];
    ob[(size_t)(ib + 16 + r) * HID + hb]      = c10[r];
    ob[(size_t)(ib + 16 + r) * HID + hb + 16] = c11[r];
  }
}

// ---------------------------------------------------------------------------
extern "C" void kernel_launch(void* const* d_in, const int* in_sizes, int n_in,
                              void* d_out, int out_size, void* d_ws, size_t ws_size,
                              hipStream_t stream) {
  const float* x  = (const float*)d_in[0];
  const float* Wq = (const float*)d_in[1];
  const float* bq = (const float*)d_in[2];
  const float* Wk = (const float*)d_in[3];
  const float* bk = (const float*)d_in[4];
  const float* Wv = (const float*)d_in[5];
  const float* bv = (const float*)d_in[6];
  float* out = (float*)d_out;

  char* ws = (char*)d_ws;
  const size_t WT = (size_t)EMB * HID * sizeof(_Float16);   // 2 MiB each
  const size_t QB = (size_t)ROWS * HID * sizeof(_Float16);  // 32 MiB each
  const size_t SB = (size_t)SEQ * SEQ * sizeof(float);      // 16 MiB (per-batch reuse)

  _Float16* wtq = (_Float16*)(ws);
  _Float16* wtk = (_Float16*)(ws + WT);
  _Float16* wtv = (_Float16*)(ws + 2 * WT);
  _Float16* q   = (_Float16*)(ws + 3 * WT);
  _Float16* k   = (_Float16*)(ws + 3 * WT + QB);
  _Float16* vt  = (_Float16*)(ws + 3 * WT + 2 * QB);
  float*    s   = (float*)   (ws + 3 * WT + 3 * QB);
  _Float16* p   = (_Float16*)(ws + 3 * WT + 3 * QB + SB);

  ha_wt_kernel<<<dim3((EMB * HID) / 256, 3), 256, 0, stream>>>(Wq, Wk, Wv, wtq, wtk, wtv);

  // 1024 m-tiles * 32 n-pairs = 32768 waves / 8 per block
  ha_qkv_kernel<<<dim3(4096), 256, 0, stream>>>(x, wtq, wtk, wtv, bq, bk, bv, q, k, vt);

  for (int b = 0; b < BATCH; ++b) {
    // 64 x 64 blocks of 32x32; 8 j-blocks per workgroup
    ha_scores_kernel<<<dim3(8, 64), 256, 0, stream>>>(q, k, s, b);
    ha_softmax_kernel<<<dim3(SEQ), 256, 0, stream>>>(s, p);
    // 64 i-blocks * 32 h-blocks = 2048 waves / 8 per block
    ha_av_kernel<<<dim3(256), 256, 0, stream>>>(p, vt, out, b);
  }
}
